// Tokenizer_8375186227382
// MI455X (gfx1250) — matrix operand order repaired and checked by reading, not compiled, not run on hardware
//
#include <hip/hip_runtime.h>
#include <hip/hip_bf16.h>

typedef __attribute__((ext_vector_type(16))) __bf16 v16bf;
typedef __attribute__((ext_vector_type(8)))  __bf16 v8bf;
typedef __attribute__((ext_vector_type(4)))  __bf16 v4bf;
typedef __attribute__((ext_vector_type(8)))  float  v8f;

#define DDIM   768
#define LDA    776          // padded bf16 row stride in LDS (2-way max bank conflict)
#define MT     128          // M tile per workgroup
#define NCHUNK 2048         // N range per workgroup
#define NBC    128          // N per WG pass (2 waves wide x 64)
#define WM     32           // wave M tile (2 row-groups of 16)
#define WN     64           // wave N tile (4 col-groups of 16)
#define THRV   100.0f

// ---------------- prep: codes fp32 -> bf16, c2[n] (inf if inactive) ----------------
__global__ __launch_bounds__(256)
void prep_codes(const float* __restrict__ codes, const unsigned char* __restrict__ active,
                __bf16* __restrict__ cb, float* __restrict__ c2) {
    __shared__ float red[256];
    const int n = blockIdx.x;
    float acc = 0.f;
    for (int j = threadIdx.x; j < DDIM; j += 256) {
        float v = codes[(size_t)n * DDIM + j];
        cb[(size_t)n * DDIM + j] = (__bf16)v;
        acc += v * v;
    }
    red[threadIdx.x] = acc;
    __syncthreads();
    for (int s = 128; s > 0; s >>= 1) {
        if (threadIdx.x < s) red[threadIdx.x] += red[threadIdx.x + s];
        __syncthreads();
    }
    if (threadIdx.x == 0) c2[n] = active[n] ? red[0] : __builtin_inff();
}

// ---------------- prep: x2[m] and init packed best keys ----------------
__global__ __launch_bounds__(256)
void prep_x(const float* __restrict__ x, float* __restrict__ x2,
            unsigned long long* __restrict__ best) {
    __shared__ float red[256];
    const int m = blockIdx.x;
    float acc = 0.f;
    for (int j = threadIdx.x; j < DDIM; j += 256) {
        float v = x[(size_t)m * DDIM + j];
        acc += v * v;
    }
    red[threadIdx.x] = acc;
    __syncthreads();
    for (int s = 128; s > 0; s >>= 1) {
        if (threadIdx.x < s) red[threadIdx.x] += red[threadIdx.x + s];
        __syncthreads();
    }
    if (threadIdx.x == 0) { x2[m] = red[0]; best[m] = ~0ull; }
}

// monotone float -> uint mapping (order-preserving under unsigned min)
__device__ __forceinline__ unsigned fmap(float f) {
    unsigned u = __float_as_uint(f);
    return (u >> 31) ? ~u : (u | 0x80000000u);
}

// ---------------- main fused GEMM + argmin ----------------
// grid: (M/128, N/2048), block 256 = 8 waves arranged 4(M) x 2(N).
// Wave tile 32M x 64N: per k32-step 8 WMMAs fed by 2 A frags (LDS) + 4 B frags
// (global/L0) -> 21.3 FLOP per fragment byte, balanced against WGP load paths.
__global__ __launch_bounds__(256)
void tok_gemm(const float* __restrict__ x, const __bf16* __restrict__ cb,
              const float* __restrict__ c2, unsigned long long* __restrict__ best) {
    extern __shared__ __bf16 As[];                 // MT x LDA bf16 (193.9 KB)
    const int m0  = blockIdx.x * MT;
    const int n0  = blockIdx.y * NCHUNK;
    const int tid = threadIdx.x;

    // Stage A tile into LDS, converting fp32 -> bf16 (float4 loads, b64 LDS stores)
    for (int i = tid; i < MT * (DDIM / 4); i += 256) {
        const int row = i / (DDIM / 4);
        const int c4  = i % (DDIM / 4);
        const float4 v = ((const float4*)(x + (size_t)(m0 + row) * DDIM))[c4];
        v4bf b4 = { (__bf16)v.x, (__bf16)v.y, (__bf16)v.z, (__bf16)v.w };
        *(v4bf*)(As + row * LDA + c4 * 4) = b4;
    }
    __syncthreads();

    const int wave  = tid >> 5;
    const int lane  = tid & 31;
    const int lhalf = lane >> 4;       // 0: lanes 0-15, 1: lanes 16-31
    const int l16   = lane & 15;
    const int mw    = wave & 3;        // M position of wave (rows mw*32 .. +32)
    const int nw    = wave >> 2;       // N position of wave (cols nw*64 .. +64)

    // running per-lane best keys: token = m0 + mw*32 + g*16 + 8*lhalf + r
    unsigned long long key[2][8];
#pragma unroll
    for (int g = 0; g < 2; ++g)
#pragma unroll
        for (int r = 0; r < 8; ++r) key[g][r] = ~0ull;

    const v8f vzero = {0.f, 0.f, 0.f, 0.f, 0.f, 0.f, 0.f, 0.f};
    const __bf16* a0 = As + (mw * WM + l16) * LDA + 8 * lhalf;  // A row base, group 0

    for (int nb = 0; nb < NCHUNK; nb += NBC) {
        v8f acc[2][4];
#pragma unroll
        for (int g = 0; g < 2; ++g)
#pragma unroll
            for (int c = 0; c < 4; ++c) acc[g][c] = vzero;

        // B base for this lane: column n = n0 + nb + nw*64 + l16 (+16 per c),
        // elems 0..7 -> K = k + 8*lhalf ; elems 8..15 -> K = k + 16 + 8*lhalf (matches A)
        const __bf16* bp = cb + (size_t)(n0 + nb + nw * WN + l16) * DDIM + 8 * lhalf;

        for (int k = 0; k < DDIM; k += 32) {
            // A fragments (ISA 16-bit A layout):
            // elems 0..7 -> K = k + 8*lhalf ; elems 8..15 -> K = k + 16 + 8*lhalf
            v16bf a[2];
#pragma unroll
            for (int g = 0; g < 2; ++g) {
                const __bf16* ap = a0 + g * 16 * LDA + k;
                v8bf alo = *(const v8bf*)(ap);
                v8bf ahi = *(const v8bf*)(ap + 16);
                a[g] = __builtin_shufflevector(alo, ahi,
                                               0, 1, 2, 3, 4, 5, 6, 7,
                                               8, 9, 10, 11, 12, 13, 14, 15);
            }
#pragma unroll
            for (int c = 0; c < 4; ++c) {
                const __bf16* bq = bp + (size_t)c * 16 * DDIM + k;
                v8bf blo = *(const v8bf*)(bq);
                v8bf bhi = *(const v8bf*)(bq + 16);
                v16bf b = __builtin_shufflevector(blo, bhi, 0, 1, 2, 3, 4, 5, 6, 7, 8, 9, 10, 11, 12, 13, 14, 15);
                acc[0][c] = __builtin_amdgcn_wmma_f32_16x16x32_bf16(
                    false, a[0], false, b, (short)0, acc[0][c], false, false);
                acc[1][c] = __builtin_amdgcn_wmma_f32_16x16x32_bf16(
                    false, a[1], false, b, (short)0, acc[1][c], false, false);
            }
        }

        // Fold: s = c2[n] - 2*dot ; x2[m] added at finalize.
        // C layout: VGPR r, lane -> M = r + 8*lhalf, N = l16.
#pragma unroll
        for (int c = 0; c < 4; ++c) {
            const int n = n0 + nb + nw * WN + 16 * c + l16;
            const float c2v = c2[n];
#pragma unroll
            for (int g = 0; g < 2; ++g)
#pragma unroll
                for (int r = 0; r < 8; ++r) {
                    float s = __builtin_fmaf(-2.0f, acc[g][c][r], c2v);
                    unsigned long long kk =
                        ((unsigned long long)fmap(s) << 32) | (unsigned)n;
                    key[g][r] = kk < key[g][r] ? kk : key[g][r];
                }
        }
    }

    // Cross-lane reduce (within each 16-lane half: same token per (g,r)),
    // then one atomic per token per wave.
#pragma unroll
    for (int g = 0; g < 2; ++g)
#pragma unroll
        for (int r = 0; r < 8; ++r) {
            unsigned long long k2 = key[g][r];
            for (int msk = 1; msk < 16; msk <<= 1) {
                unsigned long long o = __shfl_xor(k2, msk, 32);
                k2 = o < k2 ? o : k2;
            }
            if (l16 == 0) {
                const int tok = m0 + mw * WM + g * 16 + 8 * lhalf + r;
                atomicMin(&best[tok], k2);
            }
        }
}

// ---------------- finalize: unpack, threshold, write outputs ----------------
__global__ __launch_bounds__(256)
void tok_final(const unsigned long long* __restrict__ best,
               const float* __restrict__ x2, float* __restrict__ out, int M) {
    const int m = blockIdx.x * 256 + threadIdx.x;
    if (m >= M) return;
    const unsigned long long k = best[m];
    const unsigned um = (unsigned)(k >> 32);
    const unsigned n  = (unsigned)k;
    const unsigned ru = (um & 0x80000000u) ? (um ^ 0x80000000u) : ~um;
    const float d = x2[m] + __uint_as_float(ru);
    out[m]     = (d <= THRV) ? (float)n : -1.0f;   // idx_out (as float)
    out[M + m] = d;                                 // dmin
}

extern "C" void kernel_launch(void* const* d_in, const int* in_sizes, int n_in,
                              void* d_out, int out_size, void* d_ws, size_t ws_size,
                              hipStream_t stream) {
    const float* x            = (const float*)d_in[0];
    const float* codes        = (const float*)d_in[1];
    const unsigned char* actv = (const unsigned char*)d_in[2];

    const int M = in_sizes[0] / DDIM;   // 8192 tokens
    const int N = in_sizes[1] / DDIM;   // 16384 codes

    // workspace layout
    char* ws = (char*)d_ws;
    __bf16* cb = (__bf16*)ws;
    size_t off = (size_t)N * DDIM * sizeof(__bf16);        // 25,165,824
    float* c2 = (float*)(ws + off); off += (size_t)N * sizeof(float);
    float* x2 = (float*)(ws + off); off += (size_t)M * sizeof(float);
    off = (off + 7) & ~(size_t)7;
    unsigned long long* best = (unsigned long long*)(ws + off);

    prep_codes<<<N, 256, 0, stream>>>(codes, actv, cb, c2);
    prep_x<<<M, 256, 0, stream>>>(x, x2, best);

    dim3 grid(M / MT, N / NCHUNK);                         // 64 x 8
    const size_t smem = (size_t)MT * LDA * sizeof(__bf16); // 198,656 B < 320 KB/WGP
    tok_gemm<<<grid, dim3(256), smem, stream>>>(x, cb, c2, best);

    tok_final<<<(M + 255) / 256, 256, 0, stream>>>(best, x2, (float*)d_out, M);
}
